// HGConv_50964081935238
// MI455X (gfx1250) — matrix operand order repaired and checked by reading, not compiled, not run on hardware
//
#include <hip/hip_runtime.h>
#include <hip/hip_bf16.h>
#include <stdint.h>

// Problem constants (match reference)
#define B_ 8
#define M_ 4096
#define E_ 1024
#define D_ 128
#define NCAT_ 64

typedef __attribute__((ext_vector_type(16))) __bf16 v16bf;
typedef __attribute__((ext_vector_type(8)))  float  v8f;

__device__ __forceinline__ unsigned short f2bf(float f) {
    union { float f; uint32_t u; } c; c.f = f;
    uint32_t u = c.u;
    u += 0x7FFFu + ((u >> 16) & 1u);   // round-to-nearest-even
    return (unsigned short)(u >> 16);
}

__device__ __forceinline__ v8f wmma_bf16(v16bf a, v16bf b, v8f c) {
    return __builtin_amdgcn_wmma_f32_16x16x32_bf16(
        /*neg_a=*/false, a, /*neg_b=*/false, b,
        /*c_mod=*/(short)0, c, /*reuse_a=*/false, /*reuse_b=*/false);
}

// A fragment (16x32 bf16): element (row, k) = base[row*ld + k]
// ISA layout: lanes 0-15 row=lane, lanes 16-31 row=lane-16;
// v0..3: K = kh*8 + 2v ; v4..7: K = 16 + kh*8 + 2(v-4), kh = lane>>4
__device__ __forceinline__ v16bf load_frag_A(const unsigned short* base, int ld) {
    const int lane = threadIdx.x & 31;
    const int row = lane & 15, kh = lane >> 4;
    union { v16bf v; uint32_t u[8]; } r;
    const int rb = row * ld;
#pragma unroll
    for (int v = 0; v < 4; ++v)
        r.u[v] = *(const uint32_t*)&base[rb + kh * 8 + 2 * v];
#pragma unroll
    for (int v = 0; v < 4; ++v)
        r.u[4 + v] = *(const uint32_t*)&base[rb + 16 + kh * 8 + 2 * v];
    return r.v;
}

// B fragment (32x16 bf16): element (k, col) = base[col*ld + k]  (col-major staged)
// col = lane&15; v0..3: K = kh*8 + 2v ; v4..7: K = 16 + kh*8 + 2(v-4), kh = lane>>4
__device__ __forceinline__ v16bf load_frag_B(const unsigned short* base, int ld) {
    const int lane = threadIdx.x & 31;
    const int col = lane & 15, kh = lane >> 4;
    union { v16bf v; uint32_t u[8]; } r;
    const int cb = col * ld + kh * 8;
#pragma unroll
    for (int v = 0; v < 4; ++v)
        r.u[v] = *(const uint32_t*)&base[cb + 2 * v];
#pragma unroll
    for (int v = 0; v < 4; ++v)
        r.u[4 + v] = *(const uint32_t*)&base[cb + 16 + 2 * v];
    return r.v;
}

// ---------------------------------------------------------------------------
// Generic C(rows x 128) = A(rows x 128) @ W^T  (W is 128x128 row-major [n,k])
// 128 threads = 4 waves; 16-row tile per block; wave w owns cols [32w,32w+32)
// ---------------------------------------------------------------------------
template <bool OUT_BF16>
__global__ void __launch_bounds__(128)
gemm_wT_kernel(const float* __restrict__ A, const float* __restrict__ W,
               unsigned short* __restrict__ outb, float* __restrict__ outf) {
    const int m0 = blockIdx.x * 16;
    const int tid = threadIdx.x;
    const int wave = tid >> 5, lane = tid & 31;
    const int ncol = wave * 32;

    __shared__ __align__(16) unsigned short Asl[16 * 34];
    __shared__ __align__(16) unsigned short Wsl[128 * 34];

    v8f c0 = {}, c1 = {};

    for (int k0 = 0; k0 < D_; k0 += 32) {
        __syncthreads();
        // stage A tile 16x32 (natural orientation: k contiguous)
        {
            int s = tid;
#pragma unroll
            for (int it = 0; it < 4; ++it, s += 128) {
                int row = s >> 5, k = s & 31;
                Asl[row * 34 + k] = f2bf(A[(size_t)(m0 + row) * D_ + k0 + k]);
            }
        }
        // stage W tile 128x32
        for (int s = tid; s < 128 * 32; s += 128) {
            int n = s >> 5, k = s & 31;
            Wsl[n * 34 + k] = f2bf(W[n * D_ + k0 + k]);
        }
        __syncthreads();

        v16bf a  = load_frag_A(Asl, 34);
        v16bf b0 = load_frag_B(Wsl + ncol * 34, 34);
        v16bf b1 = load_frag_B(Wsl + (ncol + 16) * 34, 34);
        c0 = wmma_bf16(a, b0, c0);
        c1 = wmma_bf16(a, b1, c1);
    }

    const int n = lane & 15, half = lane >> 4;
#pragma unroll
    for (int r = 0; r < 8; ++r) {
        int m = m0 + r + half * 8;
        size_t o0 = (size_t)m * D_ + ncol + n;
        size_t o1 = o0 + 16;
        if constexpr (OUT_BF16) {
            outb[o0] = f2bf(c0[r]);
            outb[o1] = f2bf(c1[r]);
        } else {
            outf[o0] = c0[r];
            outf[o1] = c1[r];
        }
    }
}

// ---------------------------------------------------------------------------
// Fused incidence GEMMs: for a 16-edge tile of batch b,
//   edge_scores[e, d] = sum_m inc[m, e] * node_scores[m, d]
//   agg        [e, d] = sum_m inc[m, e] * node_feats [m, d]
// One shared A fragment (inc^T, binary -> exact in bf16) feeds both WMMAs.
// ---------------------------------------------------------------------------
__global__ void __launch_bounds__(128)
inc_gemm_kernel(const float* __restrict__ inc,      // (B, M, E) f32
                const unsigned short* __restrict__ ns, // (B, M, D) bf16 bits
                const float* __restrict__ nf,       // (B, M, D) f32
                float* __restrict__ es,             // (B, E, D)
                float* __restrict__ agg) {          // (B, E, D)
    const int b  = blockIdx.y;
    const int e0 = blockIdx.x * 16;
    const int tid = threadIdx.x;
    const int wave = tid >> 5, lane = tid & 31;
    const int ncol = wave * 32;

    __shared__ __align__(16) unsigned short incT[16 * 34];   // [e][m]
    __shared__ __align__(16) unsigned short nsT[128 * 34];   // [d][m]
    __shared__ __align__(16) unsigned short nfT[128 * 34];   // [d][m]

    v8f s0 = {}, s1 = {}, a0 = {}, a1 = {};

    for (int m0 = 0; m0 < M_; m0 += 32) {
        if (m0 + 32 < M_)  // hint next inc tile into cache (global_prefetch_b8)
            __builtin_prefetch(&inc[((size_t)b * M_ + (m0 + 32)) * E_ + e0], 0, 1);
        __syncthreads();
        // stage inc^T tile: 32 m x 16 e, transposed into [e][m]
        {
            int s = tid;
#pragma unroll
            for (int it = 0; it < 4; ++it, s += 128) {
                int ml = s >> 4, e = s & 15;
                float v = inc[((size_t)b * M_ + (m0 + ml)) * E_ + e0 + e];
                incT[e * 34 + ml] = f2bf(v);
            }
        }
        // stage node tiles transposed: 32 m x 128 d -> [d][m]; 8 d per slot
        for (int s = tid; s < 32 * 16; s += 128) {
            int ml = s >> 4, d8 = s & 15;
            size_t gb = ((size_t)b * M_ + (m0 + ml)) * D_ + d8 * 8;
            const unsigned short* p = ns + gb;
            const float* q = nf + gb;
#pragma unroll
            for (int j = 0; j < 8; ++j) {
                nsT[(d8 * 8 + j) * 34 + ml] = p[j];
                nfT[(d8 * 8 + j) * 34 + ml] = f2bf(q[j]);
            }
        }
        __syncthreads();

        v16bf a   = load_frag_A(incT, 34);
        v16bf bs0 = load_frag_B(nsT + ncol * 34, 34);
        v16bf bs1 = load_frag_B(nsT + (ncol + 16) * 34, 34);
        v16bf bf0 = load_frag_B(nfT + ncol * 34, 34);
        v16bf bf1 = load_frag_B(nfT + (ncol + 16) * 34, 34);
        s0 = wmma_bf16(a, bs0, s0);
        s1 = wmma_bf16(a, bs1, s1);
        a0 = wmma_bf16(a, bf0, a0);
        a1 = wmma_bf16(a, bf1, a1);
    }

    const int n = lane & 15, half = lane >> 4;
#pragma unroll
    for (int r = 0; r < 8; ++r) {
        int e = e0 + r + half * 8;
        size_t o0 = ((size_t)b * E_ + e) * D_ + ncol + n;
        size_t o1 = o0 + 16;
        es[o0]  = s0[r];
        es[o1]  = s1[r];
        agg[o0] = a0[r];
        agg[o1] = a1[r];
    }
}

// ---------------------------------------------------------------------------
// Softmax over edge axis per (b,d) column; scales agg in place: agg *= attn
// grid = B*D blocks of 256 threads; column length E = 1024 (4 per thread)
// ---------------------------------------------------------------------------
__global__ void __launch_bounds__(256)
attn_scale_kernel(const float* __restrict__ es, float* __restrict__ agg) {
    const int b = blockIdx.x >> 7;
    const int d = blockIdx.x & 127;
    const size_t base = (size_t)b * E_ * D_ + d;
    const int t = threadIdx.x;
    __shared__ float red[256];

    float x[4];
    float mx = -3.4e38f;
#pragma unroll
    for (int i = 0; i < 4; ++i) {
        x[i] = es[base + (size_t)(t + i * 256) * D_];
        mx = fmaxf(mx, x[i]);
    }
    red[t] = mx; __syncthreads();
    for (int s = 128; s > 0; s >>= 1) {
        if (t < s) red[t] = fmaxf(red[t], red[t + s]);
        __syncthreads();
    }
    mx = red[0]; __syncthreads();

    float sm = 0.f;
#pragma unroll
    for (int i = 0; i < 4; ++i) { x[i] = __expf(x[i] - mx); sm += x[i]; }
    red[t] = sm; __syncthreads();
    for (int s = 128; s > 0; s >>= 1) {
        if (t < s) red[t] += red[t + s];
        __syncthreads();
    }
    const float inv = 1.f / red[0];
#pragma unroll
    for (int i = 0; i < 4; ++i) {
        size_t off = base + (size_t)(t + i * 256) * D_;
        agg[off] *= x[i] * inv;
    }
}

// ---------------------------------------------------------------------------
// Epilogue per batch: attention pool over edges + two small FCs -> logits
// ---------------------------------------------------------------------------
__global__ void __launch_bounds__(256)
finalize_kernel(const float* __restrict__ ef,       // (B, E, D) edge_feat
                const float* __restrict__ att_w,    // (1, D)
                const float* __restrict__ pw,       // (D, D) ec_proj_w
                const float* __restrict__ pb,       // (D,)
                const float* __restrict__ fw,       // (NCAT, D)
                const float* __restrict__ fb,       // (NCAT,)
                float* __restrict__ out) {          // (B, NCAT)
    const int b = blockIdx.x, t = threadIdx.x;
    const float* efb = ef + (size_t)b * E_ * D_;
    __shared__ float wv[E_];
    __shared__ float red[256];
    __shared__ float pooled[D_];
    __shared__ float hid[D_];

    float loc[4];
    float mx = -3.4e38f;
#pragma unroll
    for (int i = 0; i < 4; ++i) {
        int e = t + i * 256;
        const float* row = efb + (size_t)e * D_;
        float s = 0.f;
        for (int d = 0; d < D_; ++d) s += row[d] * att_w[d];
        loc[i] = s; mx = fmaxf(mx, s);
    }
    red[t] = mx; __syncthreads();
    for (int s = 128; s > 0; s >>= 1) {
        if (t < s) red[t] = fmaxf(red[t], red[t + s]);
        __syncthreads();
    }
    mx = red[0]; __syncthreads();
    float sm = 0.f;
#pragma unroll
    for (int i = 0; i < 4; ++i) { loc[i] = __expf(loc[i] - mx); sm += loc[i]; }
    red[t] = sm; __syncthreads();
    for (int s = 128; s > 0; s >>= 1) {
        if (t < s) red[t] += red[t + s];
        __syncthreads();
    }
    const float inv = 1.f / red[0];
#pragma unroll
    for (int i = 0; i < 4; ++i) wv[t + i * 256] = loc[i] * inv;
    __syncthreads();

    if (t < D_) {
        float s = 0.f;
        for (int e = 0; e < E_; ++e) s += efb[(size_t)e * D_ + t] * wv[e];
        pooled[t] = s;
    }
    __syncthreads();
    if (t < D_) {
        float s = pb[t];
        for (int k = 0; k < D_; ++k) s += pooled[k] * pw[t * D_ + k];
        hid[t] = s;
    }
    __syncthreads();
    if (t < NCAT_) {
        float s = fb[t];
        for (int d = 0; d < D_; ++d) s += hid[d] * fw[t * D_ + d];
        out[b * NCAT_ + t] = s;
    }
}

// ---------------------------------------------------------------------------
extern "C" void kernel_launch(void* const* d_in, const int* in_sizes, int n_in,
                              void* d_out, int out_size, void* d_ws, size_t ws_size,
                              hipStream_t stream) {
    const float* node_feats = (const float*)d_in[0];
    const float* inc_mat    = (const float*)d_in[1];
    const float* W_att      = (const float*)d_in[2];
    const float* W_proj     = (const float*)d_in[3];
    const float* ec_att_w   = (const float*)d_in[4];
    const float* ec_proj_w  = (const float*)d_in[5];
    const float* ec_proj_b  = (const float*)d_in[6];
    const float* fc_w       = (const float*)d_in[7];
    const float* fc_b       = (const float*)d_in[8];
    float* out = (float*)d_out;

    // workspace layout
    char* ws = (char*)d_ws;
    const size_t NS_BYTES = (size_t)B_ * M_ * D_ * sizeof(unsigned short); // 8.39 MB
    const size_t ED_BYTES = (size_t)B_ * E_ * D_ * sizeof(float);          // 4.19 MB
    unsigned short* ns  = (unsigned short*)ws;                 // node_scores (bf16)
    float* es  = (float*)(ws + NS_BYTES);                      // edge_scores
    float* agg = (float*)(ws + NS_BYTES + ED_BYTES);           // agg -> agg*attn (in place)
    float* ef  = (float*)(ws + NS_BYTES + 2 * ED_BYTES);       // edge_feat

    // 1) node_scores = node_feats @ W_att^T   (bf16 out)
    gemm_wT_kernel<true><<<dim3((B_ * M_) / 16), 128, 0, stream>>>(
        node_feats, W_att, ns, nullptr);

    // 2) fused: edge_scores = inc^T @ node_scores ; agg = inc^T @ node_feats
    inc_gemm_kernel<<<dim3(E_ / 16, B_), 128, 0, stream>>>(
        inc_mat, ns, node_feats, es, agg);

    // 3) attn = softmax_e(edge_scores); agg *= attn
    attn_scale_kernel<<<dim3(B_ * D_), 256, 0, stream>>>(es, agg);

    // 4) edge_feat = (agg * attn) @ W_proj^T  (f32 out)
    gemm_wT_kernel<false><<<dim3((B_ * E_) / 16), 128, 0, stream>>>(
        agg, W_proj, nullptr, ef);

    // 5) pool + FCs -> logits
    finalize_kernel<<<dim3(B_), 256, 0, stream>>>(
        ef, ec_att_w, ec_proj_w, ec_proj_b, fc_w, fc_b, out);
}
